// dp_spatial_att_80728205295757
// MI455X (gfx1250) — matrix operand order repaired and checked by reading, not compiled, not run on hardware
//
#include <hip/hip_runtime.h>

// ---------------------------------------------------------------------------
// ST-GCN spatial attention block for MI455X (gfx1250, wave32, WMMA f16->f32)
// Fragment-native data layouts: every WMMA operand is 1-2 wide loads per lane;
// every D-fragment writeback is a packed v8h (global_store_b128) per register.
// ---------------------------------------------------------------------------
#define H_    8
#define CIN   64
#define DIM_  128
#define T_    120
#define V_    25
#define W_    3
#define DH_   16
#define U_    75               // 3*V
#define UP_   96               // U padded to 3 K-steps of 32
#define VP_   32               // V padded to 2 N-tiles of 16
#define KQK_  (DH_ * T_)       // 1920 = attention contraction length
#define KPAD_ (KQK_ + 32)      // p row with 16-halo on both ends
#define NB_   64
#define NEG_  (-9e15f)

typedef __attribute__((ext_vector_type(16))) _Float16 v16h;
typedef __attribute__((ext_vector_type(8)))  _Float16 v8h;
typedef __attribute__((ext_vector_type(8)))  float    v8f;
typedef __attribute__((ext_vector_type(4)))  unsigned int u32x4;
typedef __attribute__((ext_vector_type(8)))  int          i32x8;

static __device__ __forceinline__ v8f wmma16(v16h a, v16h b, v8f c) {
  return __builtin_amdgcn_wmma_f32_16x16x32_f16(false, a, false, b, (short)0, c,
                                                false, false);
}
// A fragment (16x32 f16): lane row = lane&15; two contiguous 8-half runs at
// k = hi*16 and k = hi*16+8 (hi = lane>>4): contiguous k = 16*hi + i, as load_b.
static __device__ __forceinline__ v16h make_a(const _Float16* run0,
                                              const _Float16* run1) {
  v8h lo = *(const v8h*)run0;
  v8h hi = *(const v8h*)run1;
  return __builtin_shufflevector(lo, hi, 0, 1, 2, 3, 4, 5, 6, 7,
                                 8, 9, 10, 11, 12, 13, 14, 15);
}
// B fragment (32x16 f16): lane col = lane&15; k = hi*16 + i (16 contiguous).
static __device__ __forceinline__ v16h load_b(const _Float16* p) {
  return *(const v16h*)p;
}
// D f32 16x16: element r -> row = r + 8*hi, col = lane&15.

// ---------------------------------------------------------------------------
__global__ void cvt_f32_to_f16(const float* __restrict__ src,
                               _Float16* __restrict__ dst, int n) {
  int i = blockIdx.x * blockDim.x + threadIdx.x;
  if (i < n) dst[i] = (_Float16)src[i];
}

// Spatial positional encoding table pe[v][c] (depends only on joint id).
__global__ void pe_kernel(float* __restrict__ pe_tab) {
  int idx = blockIdx.x * blockDim.x + threadIdx.x;
  if (idx < V_ * CIN) {
    int v = idx / CIN, c = idx % CIN;
    float freq = expf((float)(c & ~1) * (-logf(10000.f) / (float)CIN));
    float ang = (float)v * freq;
    pe_tab[idx] = ((c & 1) == 0) ? sinf(ang) : cosf(ang);
  }
}

// Structural-encoding table se[u][v][c] = se_pe[dis_matrix[u][v]][c].
__global__ void se_kernel(const float* __restrict__ se_pe,
                          const int* __restrict__ dis_matrix,
                          float* __restrict__ se_tab) {
  int idx = blockIdx.x * blockDim.x + threadIdx.x;
  if (idx < V_ * V_ * CIN) {
    int c = idx % CIN;
    int uv = idx / CIN;
    se_tab[idx] = se_pe[dis_matrix[uv] * CIN + c];
  }
}

// Fill the +-16 kidx halo of every p row with bq (zero-padded window conv ->
// the q value at an out-of-range time step is exactly the bias).
__global__ void halo_kernel(const float* __restrict__ bq,
                            _Float16* __restrict__ p16) {
  int n = blockIdx.x / H_, h = blockIdx.x % H_;
  for (int idx = threadIdx.x; idx < V_ * 32; idx += blockDim.x) {
    int v = idx / 32, j = idx % 32;
    int side = j >> 4, dh = j & 15;
    size_t row = ((size_t)(n * H_ + h) * V_ + v) * KPAD_;
    p16[row + (side ? (16 + KQK_ + dh) : dh)] = (_Float16)bq[h * DH_ + dh];
  }
}

// ---------------------------------------------------------------------------
// y = x + spatial PE + structural-encoding bias; pack y,x as f16 [n][t][v][c]
__global__ void prep_kernel(const float* __restrict__ x,
                            const float* __restrict__ pe_tab,
                            const float* __restrict__ se_tab,
                            _Float16* __restrict__ y16,
                            _Float16* __restrict__ x16) {
  int n = blockIdx.x / T_;
  int t = blockIdx.x % T_;
  int c = threadIdx.x;  // 0..63
  float xr[V_];
  size_t xbase = (((size_t)n * CIN + c) * T_ + t) * V_;
  for (int v = 0; v < V_; ++v) xr[v] = x[xbase + v];
  for (int v = 0; v < V_; ++v) {
    float s = 0.f;
    for (int u = 0; u < V_; ++u)
      s = fmaf(xr[u], se_tab[(u * V_ + v) * CIN + c], s);
    size_t o = (((size_t)n * T_ + t) * V_ + v) * CIN + c;
    y16[o] = (_Float16)(xr[v] + pe_tab[v * CIN + c] + s);
    x16[o] = (_Float16)xr[v];
  }
}

// ---------------------------------------------------------------------------
// Per (n,t): p = Wq*y + bq (layout [n][h][v][16+kidx]) ;
//            ke = (Wk*y + bk) + beta*((We*y + be)·G)  (layout [n][h][v32][kidx])
__global__ void qke_kernel(const _Float16* __restrict__ y16,
                           const _Float16* __restrict__ wq16,
                           const _Float16* __restrict__ wk16,
                           const _Float16* __restrict__ we16,
                           const float* __restrict__ bq,
                           const float* __restrict__ bk,
                           const float* __restrict__ be,
                           const float* __restrict__ G,
                           const float* __restrict__ beta,
                           _Float16* __restrict__ p16,
                           _Float16* __restrict__ ke16) {
  __shared__ __attribute__((aligned(32))) _Float16 yT[VP_][CIN];  // [v][c]
  __shared__ float kS[DIM_][VP_];
  __shared__ float eS[DIM_][VP_];
  int n = blockIdx.x / T_, t = blockIdx.x % T_;
  int wv = threadIdx.x >> 5, lane = threadIdx.x & 31;
  int lr = lane & 15, hi = (lane >> 4) & 1;
  v8h zero8 = {};
  for (int idx = threadIdx.x; idx < VP_ * (CIN / 8); idx += blockDim.x) {
    int v = idx / (CIN / 8), j = (idx % (CIN / 8)) * 8;
    v8h val = zero8;
    if (v < V_)
      val = *(const v8h*)&y16[(((size_t)n * T_ + t) * V_ + v) * CIN + j];
    *(v8h*)&yT[v][j] = val;
  }
  __syncthreads();
  const _Float16* Wt[3] = { wq16, wk16, we16 };
  for (int widx = 0; widx < 3; ++widx) {
    v8f acc0 = {}, acc1 = {};
    const _Float16* wrow = Wt[widx] + (wv * 16 + lr) * CIN;
    for (int kk = 0; kk < CIN; kk += 32) {
      v16h a = make_a(wrow + kk + hi * 16, wrow + kk + hi * 16 + 8);
      v16h b0 = load_b(&yT[lr][kk + hi * 16]);
      v16h b1 = load_b(&yT[16 + lr][kk + hi * 16]);
      acc0 = wmma16(a, b0, acc0);
      acc1 = wmma16(a, b1, acc1);
    }
    if (widx == 0) {
      // lane holds dh = 8*hi + r (contiguous) for h = wv, v = lr / 16+lr:
      // pack each accumulator into one v8h -> single b128 store.
      v8h pk0, pk1;
      for (int r = 0; r < 8; ++r) {
        float bb = bq[wv * 16 + 8 * hi + r];
        pk0[r] = (_Float16)(acc0[r] + bb);
        pk1[r] = (_Float16)(acc1[r] + bb);
      }
      size_t kofs = 16 + (size_t)t * DH_ + 8 * hi;
      size_t rowb = (size_t)(n * H_ + wv) * V_;
      *(v8h*)&p16[(rowb + lr) * KPAD_ + kofs] = pk0;
      if (16 + lr < V_)
        *(v8h*)&p16[(rowb + 16 + lr) * KPAD_ + kofs] = pk1;
    } else {
      float bias = (widx == 1) ? 0.f : 0.f;  // biases added below per-row
      (void)bias;
      for (int r = 0; r < 8; ++r) {
        int row = wv * 16 + r + 8 * hi;
        if (widx == 1) {
          kS[row][lr]      = acc0[r] + bk[row];
          kS[row][16 + lr] = acc1[r] + bk[row];
        } else {
          eS[row][lr]      = acc0[r] + be[row];
          eS[row][16 + lr] = acc1[r] + be[row];
        }
      }
    }
  }
  __syncthreads();
  float bv = beta[0];
  // 8-row d-groups share (h, contiguous dh) -> one b128 store each.
  for (int idx = threadIdx.x; idx < (DIM_ / 8) * VP_; idx += blockDim.x) {
    int dg = idx / VP_, u = idx % VP_;
    v8h pk;
    for (int j = 0; j < 8; ++j) {
      int d = dg * 8 + j;
      float val = 0.f;
      if (u < V_) {
        float eg = 0.f;
        for (int v = 0; v < V_; ++v) eg = fmaf(eS[d][v], G[v * V_ + u], eg);
        val = kS[d][u] + bv * eg;
      }
      pk[j] = (_Float16)val;
    }
    int h = (dg * 8) >> 4, dh0 = (dg * 8) & 15;
    *(v8h*)&ke16[((size_t)(n * H_ + h) * VP_ + u) * KQK_ +
                 (size_t)t * DH_ + dh0] = pk;
  }
}

// ---------------------------------------------------------------------------
// Per (n,h): S = Q(75x1920) * KE^T; mask + softmax(u) + alpha/att0.
// 10 waves = 5 M-tiles x 2 N-tiles, 60 K-steps; 3 wide loads + 1 WMMA each.
__global__ void att_kernel(const _Float16* __restrict__ p16,
                           const _Float16* __restrict__ ke16,
                           const float* __restrict__ dis_graphs,
                           const float* __restrict__ att0,
                           const float* __restrict__ alpha,
                           _Float16* __restrict__ att16) {
  __shared__ float S[80][VP_];
  int n = blockIdx.x / H_, h = blockIdx.x % H_;
  int wv = threadIdx.x >> 5, lane = threadIdx.x & 31;
  int lr = lane & 15, hi = (lane >> 4) & 1;
  int mtile = wv >> 1, ntile = wv & 1;
  int u = mtile * 16 + lr;
  int uc = (u < U_) ? u : (U_ - 1);              // rows 75..79 never consumed
  int w = uc / V_, vu = uc - w * V_;
  // temporal unfold: t -> t+(w-1) is kidx -> kidx+(w-1)*16; halo absorbs edges
  const _Float16* pa = p16 + ((size_t)(n * H_ + h) * V_ + vu) * KPAD_
                           + 16 + (w - 1) * DH_ + hi * 16;
  int v = ntile * 16 + lr;                       // rows 25..31 are zero-padded
  const _Float16* pb = ke16 + ((size_t)(n * H_ + h) * VP_ + v) * KQK_ + hi * 16;
  v8f acc = {};
  for (int kk = 0; kk < KQK_; kk += 32) {
    __builtin_prefetch(pb + kk + 512, 0, 1);     // ~16 K-steps ahead
    v16h a = make_a(pa + kk, pa + kk + 8);
    v16h b = load_b(pb + kk);
    acc = wmma16(a, b, acc);
  }
  for (int r = 0; r < 8; ++r)
    S[mtile * 16 + r + 8 * hi][ntile * 16 + lr] = acc[r];
  __syncthreads();
  size_t abase = (size_t)(n * H_ + h) * (UP_ * VP_);       // [n][h][v][u]
  if (threadIdx.x < VP_) {
    int vc = threadIdx.x;
    bool active = vc < V_;
    const float invS = 1.f / (float)KQK_;
    float col[U_];
    float inva = 0.f;
    if (active) {
      float mx = -3.0e38f;
      for (int uu = 0; uu < U_; ++uu) {
        float val = S[uu][vc] * invS;
        if (!(dis_graphs[((size_t)h * V_ + (uu % V_)) * V_ + vc] > 0.f))
          val = NEG_;
        col[uu] = val;
        mx = fmaxf(mx, val);
      }
      float sum = 0.f;
      for (int uu = 0; uu < U_; ++uu) { col[uu] = expf(col[uu] - mx); sum += col[uu]; }
      inva = alpha[0] / sum;
    }
    // whole 96-wide padded row as 12 packed v8h stores (pads write zeros)
    for (int j = 0; j < UP_ / 8; ++j) {
      v8h pk = {};
      if (active) {
        for (int e = 0; e < 8; ++e) {
          int uu = j * 8 + e;
          if (uu < U_)
            pk[e] = (_Float16)(col[uu] * inva +
                               att0[((size_t)h * V_ + (uu % V_)) * V_ + vc]);
        }
      }
      *(v8h*)&att16[abase + (size_t)vc * UP_ + j * 8] = pk;
    }
  }
}

// ---------------------------------------------------------------------------
// Per (n,t): res = Wd*x ; per h: xs_h = xwin(64x96)*att_h, out += Wo_h*xs_h.
// att tile (6KB contiguous) staged into LDS by the Tensor Data Mover.
__global__ void out_kernel(const _Float16* __restrict__ x16,
                           const _Float16* __restrict__ att16,
                           const _Float16* __restrict__ wo16,
                           const _Float16* __restrict__ wd16,
                           const float* __restrict__ b_o, const float* __restrict__ g_o,
                           const float* __restrict__ be_o, const float* __restrict__ m_o,
                           const float* __restrict__ v_o,
                           const float* __restrict__ b_d, const float* __restrict__ g_d,
                           const float* __restrict__ be_d, const float* __restrict__ m_d,
                           const float* __restrict__ v_d,
                           float* __restrict__ out) {
  __shared__ __attribute__((aligned(32))) _Float16 xS[CIN][UP_];   // A: [c][u]
  __shared__ __attribute__((aligned(32))) _Float16 attT[VP_][UP_]; // B: [v][u]
  __shared__ __attribute__((aligned(32))) _Float16 xsT[VP_][CIN];  // B: [v][c]
  __shared__ __attribute__((aligned(32))) _Float16 xcT[VP_][CIN];  // B: [v][c]
  int n = blockIdx.x / T_, t = blockIdx.x % T_;
  int wv = threadIdx.x >> 5, lane = threadIdx.x & 31;
  int lr = lane & 15, hi = (lane >> 4) & 1;
  v8h zero8 = {};
  for (int idx = threadIdx.x; idx < CIN * UP_; idx += blockDim.x) {
    int c = idx / UP_, u = idx % UP_;
    _Float16 val = (_Float16)0.f;
    if (u < U_) {
      int w = u / V_, vu = u - w * V_;
      int tt = t + w - 1;
      if (tt >= 0 && tt < T_)
        val = x16[(((size_t)n * T_ + tt) * V_ + vu) * CIN + c];
    }
    xS[c][u] = val;
  }
  for (int idx = threadIdx.x; idx < VP_ * (CIN / 8); idx += blockDim.x) {
    int v = idx / (CIN / 8), j = (idx % (CIN / 8)) * 8;
    v8h val = zero8;
    if (v < V_)
      val = *(const v8h*)&x16[(((size_t)n * T_ + t) * V_ + v) * CIN + j];
    *(v8h*)&xcT[v][j] = val;
  }
  __syncthreads();
  // residual GEMM: Wd(128x64) * x_center
  v8f r0 = {}, r1 = {};
  {
    const _Float16* wrow = wd16 + (wv * 16 + lr) * CIN;
    for (int kk = 0; kk < CIN; kk += 32) {
      v16h a = make_a(wrow + kk + hi * 16, wrow + kk + hi * 16 + 8);
      r0 = wmma16(a, load_b(&xcT[lr][kk + hi * 16]), r0);
      r1 = wmma16(a, load_b(&xcT[16 + lr][kk + hi * 16]), r1);
    }
  }
  v8f o0 = {}, o1 = {};
  size_t abase0 = (size_t)(n * H_) * (UP_ * VP_);
  unsigned ldsAtt = (unsigned)(unsigned long long)(uintptr_t)&attT[0][0];
  for (int h = 0; h < H_; ++h) {
    if (wv == 0) {
      // ---- Tensor Data Mover: 1-D tile of 3072 f16 -> LDS -------------
      unsigned long long ga = (unsigned long long)(uintptr_t)
          (att16 + abase0 + (size_t)h * (UP_ * VP_));
      u32x4 g0;
      g0[0] = 1u;                                        // count=1
      g0[1] = ldsAtt;                                    // lds_addr
      g0[2] = (unsigned)(ga & 0xffffffffull);            // global_addr lo
      g0[3] = (unsigned)((ga >> 32) & 0x01ffffffull) | (2u << 30);  // type=2
      i32x8 g1;
      const unsigned NEL = UP_ * VP_;                    // 3072 elements
      g1[0] = (int)(1u << 16);                           // data_size = 2B
      g1[1] = (int)((NEL & 0xffffu) << 16);              // tensor_dim0 lo
      g1[2] = (int)(((NEL >> 16) & 0xffffu) | (1u << 16)); // td0 hi | td1=1
      g1[3] = (int)((NEL & 0xffffu) << 16);              // tile_dim0
      g1[4] = 0;
      g1[5] = (int)NEL;                                  // tensor_dim0_stride
      g1[6] = 0;
      g1[7] = 0;
      asm volatile("tensor_load_to_lds %0, %1" :: "s"(g0), "s"(g1) : "memory");
      __builtin_amdgcn_s_wait_tensorcnt(0);
    }
    __syncthreads();
    if (wv < 4) {  // whole-wave uniform branch: EXEC all-ones inside
      v8f xa0 = {}, xa1 = {};
      for (int kk = 0; kk < UP_; kk += 32) {
        v16h a = make_a(&xS[wv * 16 + lr][kk + hi * 16],
                        &xS[wv * 16 + lr][kk + hi * 16 + 8]);
        xa0 = wmma16(a, load_b(&attT[lr][kk + hi * 16]), xa0);
        xa1 = wmma16(a, load_b(&attT[16 + lr][kk + hi * 16]), xa1);
      }
      for (int r = 0; r < 8; ++r) {
        int row = wv * 16 + r + 8 * hi;
        xsT[lr][row]      = (_Float16)xa0[r];
        xsT[16 + lr][row] = (_Float16)xa1[r];
      }
    }
    __syncthreads();
    {
      const _Float16* wrow = wo16 + (wv * 16 + lr) * (H_ * CIN) + h * CIN;
      for (int kk = 0; kk < CIN; kk += 32) {
        v16h a = make_a(wrow + kk + hi * 16, wrow + kk + hi * 16 + 8);
        o0 = wmma16(a, load_b(&xsT[lr][kk + hi * 16]), o0);
        o1 = wmma16(a, load_b(&xsT[16 + lr][kk + hi * 16]), o1);
      }
    }
    __syncthreads();
  }
  for (int r = 0; r < 8; ++r) {
    int d = wv * 16 + r + 8 * hi;
    float so = g_o[d] * rsqrtf(v_o[d] + 1e-5f);
    float to = be_o[d] - m_o[d] * so;
    float sd = g_d[d] * rsqrtf(v_d[d] + 1e-5f);
    float td = be_d[d] - m_d[d] * sd;
    size_t obase = (((size_t)n * DIM_ + d) * T_ + t) * V_;
    if (lr < V_) {
      float s = (o0[r] + b_o[d]) * so + to + (r0[r] + b_d[d]) * sd + td;
      out[obase + lr] = s > 0.f ? s : 0.1f * s;
    }
    if (16 + lr < V_) {
      float s = (o1[r] + b_o[d]) * so + to + (r1[r] + b_d[d]) * sd + td;
      out[obase + 16 + lr] = s > 0.f ? s : 0.1f * s;
    }
  }
}

// ---------------------------------------------------------------------------
extern "C" void kernel_launch(void* const* d_in, const int* in_sizes, int n_in,
                              void* d_out, int out_size, void* d_ws, size_t ws_size,
                              hipStream_t stream) {
  (void)in_sizes; (void)n_in; (void)out_size; (void)ws_size;
  const float* x      = (const float*)d_in[0];
  const float* Wq     = (const float*)d_in[1];
  const float* bq     = (const float*)d_in[2];
  const float* Wk     = (const float*)d_in[3];
  const float* bk     = (const float*)d_in[4];
  const float* We     = (const float*)d_in[5];
  const float* be     = (const float*)d_in[6];
  const float* G      = (const float*)d_in[7];
  const float* se_pe  = (const float*)d_in[8];
  const float* att0   = (const float*)d_in[9];
  const float* alpha  = (const float*)d_in[10];
  const float* beta   = (const float*)d_in[11];
  const float* Wo     = (const float*)d_in[12];
  const float* b_o    = (const float*)d_in[13];
  const float* g_o    = (const float*)d_in[14];
  const float* be_o   = (const float*)d_in[15];
  const float* m_o    = (const float*)d_in[16];
  const float* v_o    = (const float*)d_in[17];
  const float* Wd     = (const float*)d_in[18];
  const float* b_d    = (const float*)d_in[19];
  const float* g_d    = (const float*)d_in[20];
  const float* be_d   = (const float*)d_in[21];
  const float* m_d    = (const float*)d_in[22];
  const float* v_d    = (const float*)d_in[23];
  const int*   dis_matrix = (const int*)d_in[24];
  const float* dis_graphs = (const float*)d_in[25];
  float* out = (float*)d_out;

  char* ws = (char*)d_ws;
  size_t off = 0;
  auto alloc = [&](size_t bytes) -> void* {
    void* p = ws + off;
    off = (off + bytes + 255) & ~(size_t)255;
    return p;
  };
  size_t nyv  = (size_t)NB_ * T_ * V_ * CIN;          // y16 / x16
  size_t np   = (size_t)NB_ * H_ * V_ * KPAD_;        // p16 (halo rows)
  size_t nke  = (size_t)NB_ * H_ * VP_ * KQK_;        // ke16 (padded v)
  size_t natt = (size_t)NB_ * H_ * UP_ * VP_;         // att16
  _Float16* y16   = (_Float16*)alloc(nyv * 2);
  _Float16* x16   = (_Float16*)alloc(nyv * 2);
  _Float16* p16   = (_Float16*)alloc(np * 2);
  _Float16* ke16  = (_Float16*)alloc(nke * 2);
  _Float16* att16 = (_Float16*)alloc(natt * 2);
  _Float16* wq16  = (_Float16*)alloc((size_t)DIM_ * CIN * 2);
  _Float16* wk16  = (_Float16*)alloc((size_t)DIM_ * CIN * 2);
  _Float16* we16  = (_Float16*)alloc((size_t)DIM_ * CIN * 2);
  _Float16* wd16  = (_Float16*)alloc((size_t)DIM_ * CIN * 2);
  _Float16* wo16  = (_Float16*)alloc((size_t)DIM_ * H_ * CIN * 2);
  float*    pe_tab = (float*)alloc((size_t)V_ * CIN * 4);
  float*    se_tab = (float*)alloc((size_t)V_ * V_ * CIN * 4);

  cvt_f32_to_f16<<<(DIM_ * CIN + 255) / 256, 256, 0, stream>>>(Wq, wq16, DIM_ * CIN);
  cvt_f32_to_f16<<<(DIM_ * CIN + 255) / 256, 256, 0, stream>>>(Wk, wk16, DIM_ * CIN);
  cvt_f32_to_f16<<<(DIM_ * CIN + 255) / 256, 256, 0, stream>>>(We, we16, DIM_ * CIN);
  cvt_f32_to_f16<<<(DIM_ * CIN + 255) / 256, 256, 0, stream>>>(Wd, wd16, DIM_ * CIN);
  cvt_f32_to_f16<<<(DIM_ * H_ * CIN + 255) / 256, 256, 0, stream>>>(Wo, wo16, DIM_ * H_ * CIN);
  pe_kernel<<<(V_ * CIN + 255) / 256, 256, 0, stream>>>(pe_tab);
  se_kernel<<<(V_ * V_ * CIN + 255) / 256, 256, 0, stream>>>(se_pe, dis_matrix, se_tab);
  halo_kernel<<<NB_ * H_, 128, 0, stream>>>(bq, p16);

  prep_kernel<<<NB_ * T_, CIN, 0, stream>>>(x, pe_tab, se_tab, y16, x16);
  qke_kernel<<<NB_ * T_, 256, 0, stream>>>(y16, wq16, wk16, we16, bq, bk, be, G,
                                           beta, p16, ke16);
  att_kernel<<<NB_ * H_, 320, 0, stream>>>(p16, ke16, dis_graphs, att0,
                                           alpha, att16);
  out_kernel<<<NB_ * T_, 256, 0, stream>>>(x16, att16, wo16, wd16,
                                           b_o, g_o, be_o, m_o, v_o,
                                           b_d, g_d, be_d, m_d, v_d, out);
}
